// GCN_56126632624750
// MI455X (gfx1250) — matrix operand order repaired and checked by reading, not compiled, not run on hardware
//
#include <hip/hip_runtime.h>

typedef __attribute__((ext_vector_type(16))) _Float16 v16h;
typedef __attribute__((ext_vector_type(8)))  float    v8f;

// ---------------------------------------------------------------- utilities
__global__ void fill_f32(float* __restrict__ p, float v, int n) {
    int i = blockIdx.x * blockDim.x + threadIdx.x;
    if (i < n) p[i] = v;
}

// deg accumulation: deg pre-filled with 1.0f (self loop)
__global__ void deg_kernel(const int* __restrict__ dst, float* __restrict__ deg, int E) {
    int e = blockIdx.x * blockDim.x + threadIdx.x;
    if (e < E) atomicAdd(&deg[dst[e]], 1.0f);
}

// dinv = rsqrt(deg) in place (deg >= 1 always because of self loops)
__global__ void rsqrt_kernel(float* __restrict__ d, int n) {
    int i = blockIdx.x * blockDim.x + threadIdx.x;
    if (i < n) d[i] = rsqrtf(d[i]);
}

// ------------------------------------------------------- edge aggregation
// F=2 (layer 1 operates on raw 2-wide features: aggregate-then-GEMM)
__global__ void edge_agg2(const float* __restrict__ t, const int* __restrict__ src,
                          const int* __restrict__ dst, const float* __restrict__ dinv,
                          float* __restrict__ agg, int E) {
    int e = blockIdx.x * blockDim.x + threadIdx.x;
    if (e >= E) return;
    int s = src[e], d = dst[e];
    float nrm = dinv[s] * dinv[d];
    float2 v = *(const float2*)(t + (size_t)s * 2);
    atomicAdd(&agg[(size_t)d * 2 + 0], v.x * nrm);
    atomicAdd(&agg[(size_t)d * 2 + 1], v.y * nrm);
}

// F=16: 4 threads per edge, each moves a float4 (128-bit gather + 4 f32 atomics)
__global__ void edge_agg16(const float* __restrict__ t, const int* __restrict__ src,
                           const int* __restrict__ dst, const float* __restrict__ dinv,
                           float* __restrict__ agg, int E) {
    int gid = blockIdx.x * blockDim.x + threadIdx.x;
    int e = gid >> 2;
    int q = (gid & 3) << 2;
    if (e >= E) return;
    __builtin_prefetch(src + e + 2048, 0, 0);   // global_prefetch_b8
    int s = src[e], d = dst[e];
    float nrm = dinv[s] * dinv[d];
    float4 v = *(const float4*)(t + (size_t)s * 16 + q);
    float* a = agg + (size_t)d * 16 + q;
    atomicAdd(a + 0, v.x * nrm);
    atomicAdd(a + 1, v.y * nrm);
    atomicAdd(a + 2, v.z * nrm);
    atomicAdd(a + 3, v.w * nrm);
}

// F=3 (layer 4 projects 16->3 first, then aggregates only 3 floats/edge)
__global__ void edge_agg3(const float* __restrict__ t, const int* __restrict__ src,
                          const int* __restrict__ dst, const float* __restrict__ dinv,
                          float* __restrict__ agg, int E) {
    int e = blockIdx.x * blockDim.x + threadIdx.x;
    if (e >= E) return;
    int s = src[e], d = dst[e];
    float nrm = dinv[s] * dinv[d];
    const float* ts = t + (size_t)s * 3;
    float* a = agg + (size_t)d * 3;
    atomicAdd(a + 0, ts[0] * nrm);
    atomicAdd(a + 1, ts[1] * nrm);
    atomicAdd(a + 2, ts[2] * nrm);
}

// ------------------------------------------------------------ dense parts
// Layer 1 fused: h1 = relu((aggx + x*dinv^2) @ W1 + b1), W1 is [2,16]
__global__ void l1_fused(const float* __restrict__ aggx, const float* __restrict__ x,
                         const float* __restrict__ dinv, const float* __restrict__ W1,
                         const float* __restrict__ b1, float* __restrict__ h1, int N) {
    int i = blockIdx.x * blockDim.x + threadIdx.x;
    if (i >= N * 16) return;
    int n = i >> 4, f = i & 15;
    float di2 = dinv[n] * dinv[n];
    float a0 = aggx[(size_t)n * 2 + 0] + x[(size_t)n * 2 + 0] * di2;
    float a1 = aggx[(size_t)n * 2 + 1] + x[(size_t)n * 2 + 1] * di2;
    float v = fmaf(a0, W1[f], fmaf(a1, W1[16 + f], b1[f]));
    h1[i] = fmaxf(v, 0.0f);
}

// Y[N,16] = X[N,16] @ W[16,16] via v_wmma_f32_16x16x32_f16 (K padded 16->32).
// One wave computes a 16x16 output tile. blockDim must be a multiple of 32.
// A layout: lanes 0-15 hold K=0..7 of row M=lane in v0..v3 (f16 pairs),
//           lanes 16-31 hold K=8..15 of row M=lane-16; v4..v7 (K=16..31) = 0 pad.
// B layout: lanes 0-15 hold K=0..7 of column N=lane, lanes 16-31 hold K=8..15 of N=lane-16; v4..v7 (K=16..31) = 0 pad.
// D layout: VGPR r -> row M = r + 8*(lane>=16), col N = lane&15.
// Fast path (full tile, uniform per wave): unguarded b128 loads + coalesced stores.
__global__ void gemm16_wmma(const float* __restrict__ X, const float* __restrict__ W,
                            float* __restrict__ Y, int N) {
    int lane = threadIdx.x & 31;
    int wave = threadIdx.x >> 5;
    int rowBase = (blockIdx.x * (blockDim.x >> 5) + wave) * 16;
    int m  = lane & 15;
    int hi = lane >> 4;          // 0: K 0..7 / D rows 0..7; 1: K 8..15 / D rows 8..15
    int row = rowBase + m;
    bool full = (rowBase + 16 <= N);   // uniform across the wave

    v16h a = {};
    if (full) {
        const float4* p = (const float4*)(X + (size_t)row * 16 + hi * 8);
        float4 v0 = p[0];                       // global_load_b128
        float4 v1 = p[1];                       // global_load_b128
        a[0] = (_Float16)v0.x; a[1] = (_Float16)v0.y;
        a[2] = (_Float16)v0.z; a[3] = (_Float16)v0.w;
        a[4] = (_Float16)v1.x; a[5] = (_Float16)v1.y;
        a[6] = (_Float16)v1.z; a[7] = (_Float16)v1.w;
    } else if (row < N) {
        const float* xr = X + (size_t)row * 16 + hi * 8;
        #pragma unroll
        for (int j = 0; j < 8; ++j) a[j] = (_Float16)xr[j];
    }
    v16h b = {};
    {
        #pragma unroll
        for (int k = 0; k < 8; ++k) b[k] = (_Float16)W[(hi * 8 + k) * 16 + m];
    }
    v8f c = {};
    // EXEC is all-ones here (no early returns; guards above are re-converged ifs)
    c = __builtin_amdgcn_wmma_f32_16x16x32_f16(false, a, false, b, (short)0, c,
                                               false, false);
    if (full) {
        float* yr = Y + (size_t)(rowBase + hi * 8) * 16 + m;
        #pragma unroll
        for (int r = 0; r < 8; ++r) yr[r * 16] = c[r];   // coalesced, unguarded
    } else {
        #pragma unroll
        for (int r = 0; r < 8; ++r) {
            int mm = rowBase + r + hi * 8;
            if (mm < N) Y[(size_t)mm * 16 + m] = c[r];
        }
    }
}

// epilogue for 16-wide layers: out = agg + t*dinv^2 + b (+res) (relu)
__global__ void epilogue16(const float* __restrict__ agg, const float* __restrict__ t,
                           const float* __restrict__ dinv, const float* __restrict__ b,
                           const float* __restrict__ res, float* __restrict__ out,
                           int N, int doRelu) {
    int i = blockIdx.x * blockDim.x + threadIdx.x;
    if (i >= N * 16) return;
    int n = i >> 4, f = i & 15;
    float di = dinv[n];
    float v = agg[i] + t[i] * di * di + b[f];
    if (res) v += res[i];
    if (doRelu) v = fmaxf(v, 0.0f);
    out[i] = v;
}

// t4[N,3] = h3[N,16] @ W4[16,3]
__global__ void gemm16x3(const float* __restrict__ h3, const float* __restrict__ W4,
                         float* __restrict__ t4, int N) {
    int i = blockIdx.x * blockDim.x + threadIdx.x;
    if (i >= N * 3) return;
    int n = i / 3, f = i - n * 3;
    const float* hr = h3 + (size_t)n * 16;
    float acc = 0.0f;
    #pragma unroll
    for (int k = 0; k < 16; ++k) acc = fmaf(hr[k], W4[k * 3 + f], acc);
    t4[i] = acc;
}

// final epilogue: out = agg + t*dinv^2 + b4
__global__ void epilogue3(const float* __restrict__ agg, const float* __restrict__ t,
                          const float* __restrict__ dinv, const float* __restrict__ b,
                          float* __restrict__ out, int N) {
    int i = blockIdx.x * blockDim.x + threadIdx.x;
    if (i >= N * 3) return;
    int n = i / 3, f = i - n * 3;
    float di = dinv[n];
    out[i] = agg[i] + t[i] * di * di + b[f];
}

// ---------------------------------------------------------------- launcher
extern "C" void kernel_launch(void* const* d_in, const int* in_sizes, int n_in,
                              void* d_out, int out_size, void* d_ws, size_t ws_size,
                              hipStream_t stream) {
    const float* x  = (const float*)d_in[0];
    const int*   ei = (const int*)d_in[1];
    const float* W1 = (const float*)d_in[2];
    const float* b1 = (const float*)d_in[3];
    const float* W2 = (const float*)d_in[4];
    const float* b2 = (const float*)d_in[5];
    const float* W3 = (const float*)d_in[6];
    const float* b3 = (const float*)d_in[7];
    const float* W4 = (const float*)d_in[8];
    const float* b4 = (const float*)d_in[9];

    const int N = in_sizes[0] / 2;      // x is [N,2]
    const int E = in_sizes[1] / 2;      // edge_index is [2,E] row-major
    const int* src = ei;
    const int* dst = ei + E;

    float* ws   = (float*)d_ws;         // needs 67*N floats (~40 MB)
    float* dinv = ws;                   // N
    float* aggx = ws + (size_t)1  * N;  // 2N
    float* h1   = ws + (size_t)3  * N;  // 16N (also residual for layer 3)
    float* tbuf = ws + (size_t)19 * N;  // 16N (t2 / t3 / t4)
    float* aggb = ws + (size_t)35 * N;  // 16N (agg2 / agg3 / agg4)
    float* hbuf = ws + (size_t)51 * N;  // 16N (h2 then h3)
    float* out  = (float*)d_out;

    const int B = 256;
    auto cdiv = [](long long a, long long b) { return (int)((a + b - 1) / b); };

    // GEMM launch geometry: 5 waves/block (160 thr) -> 80 rows/block.
    // N=150000 -> 1875 blocks, every wave a full 16-row tile (no tail).
    const int GB = 160;                 // 5 waves
    const int gemmGrid = cdiv(N, 80);

    // ---- symmetric normalization: deg = 1 + count(dst); dinv = rsqrt(deg)
    fill_f32<<<cdiv(N, B), B, 0, stream>>>(dinv, 1.0f, N);
    deg_kernel<<<cdiv(E, B), B, 0, stream>>>(dst, dinv, E);
    rsqrt_kernel<<<cdiv(N, B), B, 0, stream>>>(dinv, N);

    // ---- layer 1: aggregate raw x (2 floats/edge), fused GEMM+bias+ReLU
    fill_f32<<<cdiv(2LL * N, B), B, 0, stream>>>(aggx, 0.0f, 2 * N);
    edge_agg2<<<cdiv(E, B), B, 0, stream>>>(x, src, dst, dinv, aggx, E);
    l1_fused<<<cdiv(16LL * N, B), B, 0, stream>>>(aggx, x, dinv, W1, b1, h1, N);

    // ---- layer 2: t2 = h1@W2 (WMMA), aggregate, +b2
    gemm16_wmma<<<gemmGrid, GB, 0, stream>>>(h1, W2, tbuf, N);
    fill_f32<<<cdiv(16LL * N, B), B, 0, stream>>>(aggb, 0.0f, 16 * N);
    edge_agg16<<<cdiv(4LL * E, B), B, 0, stream>>>(tbuf, src, dst, dinv, aggb, E);
    epilogue16<<<cdiv(16LL * N, B), B, 0, stream>>>(aggb, tbuf, dinv, b2, nullptr,
                                                    hbuf, N, 0);

    // ---- layer 3: t3 = h2@W3 (WMMA), aggregate, +b3 +h1 residual, ReLU
    gemm16_wmma<<<gemmGrid, GB, 0, stream>>>(hbuf, W3, tbuf, N);
    fill_f32<<<cdiv(16LL * N, B), B, 0, stream>>>(aggb, 0.0f, 16 * N);
    edge_agg16<<<cdiv(4LL * E, B), B, 0, stream>>>(tbuf, src, dst, dinv, aggb, E);
    epilogue16<<<cdiv(16LL * N, B), B, 0, stream>>>(aggb, tbuf, dinv, b3, h1,
                                                    hbuf, N, 1);

    // ---- layer 4: project 16->3 first, aggregate only 3 floats/edge, +b4
    gemm16x3<<<cdiv(3LL * N, B), B, 0, stream>>>(hbuf, W4, tbuf, N);
    fill_f32<<<cdiv(3LL * N, B), B, 0, stream>>>(aggb, 0.0f, 3 * N);
    edge_agg3<<<cdiv(E, B), B, 0, stream>>>(tbuf, src, dst, dinv, aggb, E);
    epilogue3<<<cdiv(3LL * N, B), B, 0, stream>>>(aggb, tbuf, dinv, b4, out, N);
}
